// MultiHeadAttention_71605694759496
// MI455X (gfx1250) — matrix operand order repaired and checked by reading, not compiled, not run on hardware
//
#include <hip/hip_runtime.h>
#include <hip/hip_bf16.h>
#include <math.h>

typedef __attribute__((ext_vector_type(16))) __bf16 v16bf;
typedef __attribute__((ext_vector_type(8)))  __bf16 v8bf;
typedef __attribute__((ext_vector_type(4)))  __bf16 v4bf;
typedef __attribute__((ext_vector_type(8)))  float  v8f;
typedef __attribute__((ext_vector_type(4)))  float  v4f;

#define S_LEN 2048
#define D_MODEL 1024
#define N_HEADS 16
#define D_HEAD 64
#define M_ROWS (4 * S_LEN)   // B*S = 8192

// ---------------------------------------------------------------------------
// WMMA wrapper: D = A(16x32 bf16) * B(32x16 bf16) + C(16x16 f32)
// ---------------------------------------------------------------------------
__device__ __forceinline__ v8f wmma_bf16(v16bf a, v16bf b, v8f c) {
  return __builtin_amdgcn_wmma_f32_16x16x32_bf16(
      /*neg_a=*/false, a, /*neg_b=*/false, b,
      /*c_mod=*/(short)0, c, /*reuse_a=*/false, /*reuse_b=*/false);
}

// A-fragment (16x32, MxK), source row-major with contraction contiguous.
// Lanes 0-15 hold row M=lane, K = {0..7, 16..23}; lanes 16-31: K = {8..15, 24..31}.
__device__ __forceinline__ v16bf frag_a(const __bf16* base, int row0,
                                        int stride, int cbase) {
  unsigned lane = threadIdx.x & 31u;
  const __bf16* p = base + (size_t)(row0 + (lane & 15u)) * stride + cbase +
                    ((lane >> 4) << 3);
  union { v16bf v; v8bf h[2]; } u;
  u.h[0] = *(const v8bf*)p;
  u.h[1] = *(const v8bf*)(p + 16);
  return u.v;
}

// B-fragment (32x16, KxN), source laid out rows=N, cols=K (contraction
// contiguous per lane). Lane holds column N=lane%16, K = {0..7, 16..23} (lanes 0-15) / {8..15, 24..31} (lanes 16-31).
__device__ __forceinline__ v16bf frag_b(const __bf16* base, int col0,
                                        int stride, int cbase) {
  unsigned lane = threadIdx.x & 31u;
  const __bf16* p = base + (size_t)(col0 + (lane & 15u)) * stride + cbase +
                    ((lane >> 4) << 3);
  union { v16bf v; v8bf h[2]; } u;
  u.h[0] = *(const v8bf*)p;
  u.h[1] = *(const v8bf*)(p + 16);
  return u.v;
}

// 4-element load with conversion to bf16 (fp32 source) or passthrough (bf16).
__device__ __forceinline__ v4bf load4_bf(const float* p) {
  v4f x = *(const v4f*)p;
  v4bf y;
  y[0] = (__bf16)x[0]; y[1] = (__bf16)x[1];
  y[2] = (__bf16)x[2]; y[3] = (__bf16)x[3];
  return y;
}
__device__ __forceinline__ v4bf load4_bf(const __bf16* p) {
  return *(const v4bf*)p;
}

// ---------------------------------------------------------------------------
// GEMM: OUT[m,n] = sum_k X[m,k] * W[n,k] + bias[n]   (torch Linear: x @ W^T + b)
// 256 threads (8 waves), 128x128 tile, k-step 64 (16 WMMAs per stage).
// ---------------------------------------------------------------------------
template <typename TIN, typename TOUT>
__global__ __launch_bounds__(256) void gemm_xwT(const TIN* __restrict__ X,
                                                const float* __restrict__ W,
                                                const float* __restrict__ bias,
                                                TOUT* __restrict__ OUT) {
  const int K = D_MODEL, N = D_MODEL;
  const int LSTR = 72;  // padded bf16 row stride (144B: 16B-aligned, conflict-free)
  __shared__ __align__(16) __bf16 At[128 * LSTR];
  __shared__ __align__(16) __bf16 Bt[128 * LSTR];

  int tid = threadIdx.x;
  int lane = tid & 31, wave = tid >> 5;
  int ll = lane & 15;
  int m0 = blockIdx.x * 128, n0 = blockIdx.y * 128;
  int wm = (wave & 1) * 64, wn = (wave >> 1) * 32;   // wave tile: 64x32

  v8f acc[4][2] = {};

  for (int k0 = 0; k0 < K; k0 += 64) {
    __syncthreads();
    // Stage 128x64 tiles of X and W into LDS as bf16 (vectorized).
    for (int u = tid; u < 2048; u += 256) {
      int r = u >> 4, c = (u & 15) << 2;
      *(v4bf*)&At[r * LSTR + c] = load4_bf(X + (size_t)(m0 + r) * K + k0 + c);
      *(v4bf*)&Bt[r * LSTR + c] = load4_bf(W + (size_t)(n0 + r) * K + k0 + c);
    }
    __syncthreads();

#pragma unroll
    for (int kk = 0; kk < 64; kk += 32) {
      v16bf b0 = frag_b(Bt, wn, LSTR, kk);
      v16bf b1 = frag_b(Bt, wn + 16, LSTR, kk);
#pragma unroll
      for (int r = 0; r < 4; ++r) {
        v16bf a = frag_a(At, wm + r * 16, LSTR, kk);
        acc[r][0] = wmma_bf16(a, b0, acc[r][0]);
        acc[r][1] = wmma_bf16(a, b1, acc[r][1]);
      }
    }
  }

  // Epilogue: C/D layout -> VGPR e holds M = e + (lane>=16 ? 8 : 0), N = lane%16
  int qhi = (lane >> 4) * 8;
#pragma unroll
  for (int r = 0; r < 4; ++r) {
#pragma unroll
    for (int c = 0; c < 2; ++c) {
      int col = n0 + wn + c * 16 + ll;
      float bb = bias[col];
#pragma unroll
      for (int e = 0; e < 8; ++e) {
        int row = m0 + wm + r * 16 + e + qhi;
        OUT[(size_t)row * N + col] = (TOUT)(acc[r][c][e] + bb);
      }
    }
  }
}

// ---------------------------------------------------------------------------
// Flash attention: one workgroup = (b, h, 64-query block); 4 waves, each wave
// owns 16 queries x 64 (d_k) output. Streams 32-key blocks, online softmax.
// K tile staged with CDNA5 async global->LDS DMA (ASYNCcnt-tracked).
// ---------------------------------------------------------------------------
__global__ __launch_bounds__(128) void attn_flash(const __bf16* __restrict__ Qb,
                                                  const __bf16* __restrict__ Kb,
                                                  const __bf16* __restrict__ Vb,
                                                  const int* __restrict__ mask,
                                                  __bf16* __restrict__ Ctx) {
  const int D = D_MODEL;
  const int KSTR = 72;  // padded stride for K/V tiles (144B, conflict-free)
  const int PSTR = 40;  // padded stride for P scratch

  __shared__ __align__(16) __bf16 Kt[32 * KSTR];       // (key, d_k) row-major
  __shared__ __align__(16) __bf16 Vt[64 * KSTR];       // (d_k, key): V transposed
  __shared__ __align__(16) __bf16 Pw[4][16 * PSTR];    // per-wave P scratch

  int qblk = blockIdx.x * 64;
  int h = blockIdx.y;
  int b = blockIdx.z;
  int tid = threadIdx.x;
  int lane = tid & 31, wave = tid >> 5;
  int ll = lane & 15;
  int qhi = (lane >> 4) * 8;

  const float NEG_INF = -__builtin_inff();

  // Preload this wave's Q fragments (16 queries x 64 d_k = two K=32 fragments)
  int qrow0 = qblk + wave * 16;
  const __bf16* Qbase = Qb + (size_t)b * S_LEN * D + (size_t)h * D_HEAD;
  v16bf aq0 = frag_a(Qbase, qrow0, D, 0);
  v16bf aq1 = frag_a(Qbase, qrow0, D, 32);

  v8f acc[4] = {};
  float mrow[8], lrow[8];
#pragma unroll
  for (int r = 0; r < 8; ++r) { mrow[r] = NEG_INF; lrow[r] = 0.0f; }

  const int* mbase = mask + (size_t)b * S_LEN * S_LEN;
  const __bf16* Khead = Kb + (size_t)b * S_LEN * D + (size_t)h * D_HEAD;
  const __bf16* Vhead = Vb + (size_t)b * S_LEN * D + (size_t)h * D_HEAD;
  __bf16* Pme = Pw[wave];

  for (int kb = 0; kb < S_LEN; kb += 32) {
    __syncthreads();
    // Stage 32-key block. K tile: async DMA straight into LDS (no VGPR
    // round-trip). V tile: through VGPRs, transposed into LDS.
    for (int u = tid; u < 256; u += 128) {
      int kr = u >> 3, ch = (u & 7) << 3;
      unsigned lds_off = (unsigned)(unsigned long long)&Kt[kr * KSTR + ch];
      unsigned long long ga =
          (unsigned long long)(Khead + (size_t)(kb + kr) * D + ch);
      asm volatile("global_load_async_to_lds_b128 %0, %1, off"
                   :: "v"(lds_off), "v"(ga) : "memory");
      v8bf dv = *(const v8bf*)(Vhead + (size_t)(kb + kr) * D + ch);
#pragma unroll
      for (int i = 0; i < 8; ++i) Vt[(ch + i) * KSTR + kr] = dv[i];
      if (kb + 32 < S_LEN)
        __builtin_prefetch(Vhead + (size_t)(kb + 32 + kr) * D + ch, 0, 0);
    }
    asm volatile("s_wait_asynccnt 0x0" ::: "memory");
    __syncthreads();

    // Scores S = Q * K^T for 16 queries x 32 keys (two 16x16 tiles).
    v8f s0 = {}, s1 = {};
    {
      v16bf bk0 = frag_b(Kt, 0, KSTR, 0);
      v16bf bk1 = frag_b(Kt, 16, KSTR, 0);
      s0 = wmma_bf16(aq0, bk0, s0);
      s1 = wmma_bf16(aq0, bk1, s1);
      bk0 = frag_b(Kt, 0, KSTR, 32);
      bk1 = frag_b(Kt, 16, KSTR, 32);
      s0 = wmma_bf16(aq1, bk0, s0);
      s1 = wmma_bf16(aq1, bk1, s1);
    }

    // Online softmax per query row; write P (bf16) to per-wave LDS scratch.
#pragma unroll
    for (int r = 0; r < 8; ++r) {
      int qi = qrow0 + r + qhi;
      const int* mr = mbase + (size_t)qi * S_LEN + kb;
      float x0 = s0[r] * 0.125f;  // 1/sqrt(64)
      float x1 = s1[r] * 0.125f;
      if (mr[ll] == 0)      x0 = NEG_INF;
      if (mr[16 + ll] == 0) x1 = NEG_INF;
      float mx = fmaxf(x0, x1);
#pragma unroll
      for (int off = 8; off >= 1; off >>= 1)
        mx = fmaxf(mx, __shfl_xor(mx, off, 16));
      float mnew = fmaxf(mrow[r], mx);
      float alpha = (mnew == mrow[r]) ? 1.0f : __expf(mrow[r] - mnew);
      float p0 = __expf(x0 - mnew);
      float p1 = __expf(x1 - mnew);
      float sum = p0 + p1;
#pragma unroll
      for (int off = 8; off >= 1; off >>= 1) sum += __shfl_xor(sum, off, 16);
      lrow[r] = lrow[r] * alpha + sum;
      mrow[r] = mnew;
#pragma unroll
      for (int t = 0; t < 4; ++t) acc[t][r] *= alpha;
      Pme[(r + qhi) * PSTR + ll] = (__bf16)p0;
      Pme[(r + qhi) * PSTR + 16 + ll] = (__bf16)p1;
    }

    // acc += P(16x32) * V(32x64); V fragments from transposed LDS tile.
    v16bf pa = frag_a(Pme, 0, PSTR, 0);
#pragma unroll
    for (int t = 0; t < 4; ++t) {
      v16bf bv = frag_b(Vt, t * 16, KSTR, 0);
      acc[t] = wmma_bf16(pa, bv, acc[t]);
    }
  }

  // Normalize and store context (bf16) for the output projection.
  __bf16* Cbase = Ctx + (size_t)b * S_LEN * D + (size_t)h * D_HEAD;
#pragma unroll
  for (int r = 0; r < 8; ++r) {
    int qi = qrow0 + r + qhi;
    float inv = 1.0f / lrow[r];
#pragma unroll
    for (int t = 0; t < 4; ++t)
      Cbase[(size_t)qi * D + t * 16 + ll] = (__bf16)(acc[t][r] * inv);
  }
}

// ---------------------------------------------------------------------------
extern "C" void kernel_launch(void* const* d_in, const int* in_sizes, int n_in,
                              void* d_out, int out_size, void* d_ws,
                              size_t ws_size, hipStream_t stream) {
  (void)in_sizes; (void)n_in; (void)out_size; (void)ws_size;
  const float* q    = (const float*)d_in[0];
  const float* k    = (const float*)d_in[1];
  const float* v    = (const float*)d_in[2];
  const int*   mask = (const int*)d_in[3];
  const float* Wq = (const float*)d_in[4];
  const float* bq = (const float*)d_in[5];
  const float* Wk = (const float*)d_in[6];
  const float* bk = (const float*)d_in[7];
  const float* Wv = (const float*)d_in[8];
  const float* bv = (const float*)d_in[9];
  const float* Wo = (const float*)d_in[10];
  const float* bo = (const float*)d_in[11];

  const size_t MD = (size_t)M_ROWS * D_MODEL;  // 8192*1024
  __bf16* Qb  = (__bf16*)d_ws;
  __bf16* Kb  = Qb + MD;
  __bf16* Vb  = Kb + MD;
  __bf16* Ctx = Vb + MD;

  dim3 gg(M_ROWS / 128, D_MODEL / 128);  // (64, 8)
  gemm_xwT<float, __bf16><<<gg, 256, 0, stream>>>(q, Wq, bq, Qb);
  gemm_xwT<float, __bf16><<<gg, 256, 0, stream>>>(k, Wk, bk, Kb);
  gemm_xwT<float, __bf16><<<gg, 256, 0, stream>>>(v, Wv, bv, Vb);

  dim3 ga(S_LEN / 64, N_HEADS, 4);  // (32, 16, 4)
  attn_flash<<<ga, 128, 0, stream>>>(Qb, Kb, Vb, mask, Ctx);

  gemm_xwT<__bf16, float><<<gg, 256, 0, stream>>>(Ctx, Wo, bo, (float*)d_out);
}
